// StyleConv_55989193671037
// MI455X (gfx1250) — matrix operand order repaired and checked by reading, not compiled, not run on hardware
//
#include <hip/hip_runtime.h>
#include <hip/hip_bf16.h>

// ---------------------------------------------------------------------------
// StyleConv (StyleGAN2 modulated conv) for gfx1250 / MI455X.
// out[b,co] = demod[b,co] * conv2d( s[b,ci]*x[b,ci], weight ) + bias[co]
// Heavy conv: implicit GEMM on v_wmma_f32_16x16x32_bf16 with double-buffered
// GLOBAL_LOAD_ASYNC_TO_LDS staging (ASYNCcnt pipeline).
// ---------------------------------------------------------------------------

typedef __attribute__((ext_vector_type(16))) __bf16 v16bf;
typedef __attribute__((ext_vector_type(8)))  float  v8f;
typedef int v4i_t __attribute__((vector_size(16)));

union Frag16 {
    uint4  u[2];
    v16bf  v;
};

#if defined(__AMDGCN__) && __has_builtin(__builtin_amdgcn_global_load_async_to_lds_b128)
#define HAS_ASYNC_LDS 1
#else
#define HAS_ASYNC_LDS 0
#endif

#if HAS_ASYNC_LDS
#if __has_builtin(__builtin_amdgcn_s_wait_asynccnt)
#define ASYNC_WAIT(n) __builtin_amdgcn_s_wait_asynccnt(n)
#else
#define ASYNC_WAIT(n) asm volatile("s_wait_asynccnt %0" ::"n"(n) : "memory")
#endif
#else
#define ASYNC_WAIT(n) ((void)0)
#endif

// Copy 16 contiguous bytes global -> LDS, per lane.
__device__ __forceinline__ void copy_b128(const unsigned short* g, unsigned short* l) {
#if HAS_ASYNC_LDS
    __builtin_amdgcn_global_load_async_to_lds_b128(
        (__attribute__((address_space(1))) v4i_t*)g,
        (__attribute__((address_space(3))) v4i_t*)l, 0, 0);
#else
    *(uint4*)l = *(const uint4*)g;
#endif
}

__device__ __forceinline__ unsigned short f32_to_bf16(float f) {
    unsigned int u = __float_as_uint(f);
    unsigned int r = u + 0x7FFFu + ((u >> 16) & 1u);   // round-to-nearest-even
    return (unsigned short)(r >> 16);
}

// ---------------------------------------------------------------------------
// 0) zero-fill the haloed activation buffer (uint4 granularity)
// ---------------------------------------------------------------------------
__global__ __launch_bounds__(256) void zero_kernel(uint4* __restrict__ p) {
    const uint4 z = {0u, 0u, 0u, 0u};
    p[(size_t)blockIdx.x * 256 + threadIdx.x] = z;
}

// ---------------------------------------------------------------------------
// 1) s[b,cin] = mod_b[cin] + dot(style[b,:], mod_w[cin,:])       (8 x 512)
// ---------------------------------------------------------------------------
__global__ __launch_bounds__(256) void style_linear_kernel(
    const float* __restrict__ style,   // [8,512]
    const float* __restrict__ mod_w,   // [512,512]
    const float* __restrict__ mod_b,   // [512]
    float* __restrict__ s)             // [8,512]
{
    __shared__ float st[512];
    const int b = blockIdx.x;
    const int t = threadIdx.x;
    st[t]       = style[b * 512 + t];
    st[t + 256] = style[b * 512 + t + 256];
    __syncthreads();
    #pragma unroll
    for (int i = 0; i < 2; ++i) {
        const int cin = i * 256 + t;
        const float* mw = mod_w + (size_t)cin * 512;
        float acc = mod_b[cin];
        for (int f = 0; f < 512; ++f) acc += st[f] * mw[f];
        s[b * 512 + cin] = acc;
    }
}

// ---------------------------------------------------------------------------
// 2) demod[b,co] = rsqrt( sum_ci s[b,ci]^2 * sum_kk w[co,ci,kk]^2 + 1e-8 )
// ---------------------------------------------------------------------------
__global__ __launch_bounds__(256) void demod_kernel(
    const float* __restrict__ weight,  // [512,512,3,3]
    const float* __restrict__ s,       // [8,512]
    float* __restrict__ demod)         // [8,512]
{
    const int b  = blockIdx.y;
    const int co = blockIdx.x * 256 + threadIdx.x;
    const float* wrow = weight + (size_t)co * 4608;
    const float* srow = s + b * 512;
    float acc = 1e-8f;
    for (int ci = 0; ci < 512; ++ci) {
        float w2 = 0.f;
        #pragma unroll
        for (int j = 0; j < 9; ++j) { float w = wrow[ci * 9 + j]; w2 += w * w; }
        const float sv = srow[ci];
        acc += sv * sv * w2;
    }
    demod[b * 512 + co] = rsqrtf(acc);
}

// ---------------------------------------------------------------------------
// 3) weight -> bf16, pre-swizzled into the exact LDS A-tile image:
//    wt[kk][ciBlk(16)][coBlk(4)] -> 8KB tile laid out [g(4)][m(128)][k%8(8)]
//    so the conv kernel's A staging is a pure linear async copy.
// ---------------------------------------------------------------------------
__global__ __launch_bounds__(256) void wtr_kernel(
    const float* __restrict__ weight,      // [co,ci,kh,kw]
    unsigned short* __restrict__ wt)       // [9*16*4*4096] bf16
{
    const int idx = blockIdx.x * 256 + threadIdx.x;   // < 9*512*512
    const int j   = idx & 7;
    const int m   = (idx >> 3) & 127;
    const int g   = (idx >> 10) & 3;
    const int coB = (idx >> 12) & 3;
    const int cb  = (idx >> 14) & 15;
    const int kk  = idx >> 18;
    const int ci  = cb * 32 + g * 8 + j;
    const int co  = coB * 128 + m;
    wt[idx] = f32_to_bf16(weight[((size_t)co * 512 + ci) * 9 + kk]);
}

// ---------------------------------------------------------------------------
// 4) xs[b][h+1][w+1][ci] = bf16( s[b,ci] * x[b,ci,h,w] )  (NHWC + zero halo)
// ---------------------------------------------------------------------------
__global__ __launch_bounds__(256) void xs_kernel(
    const float* __restrict__ x,       // [8,512,64,64]
    const float* __restrict__ s,       // [8,512]
    unsigned short* __restrict__ xs)   // [8,66,66,512] bf16 (halo pre-zeroed)
{
    __shared__ float tile[64 * 65];    // pad to kill bank conflicts
    const int b   = blockIdx.z;
    const int h   = blockIdx.y;
    const int ci0 = blockIdx.x * 64;
    const int t   = threadIdx.x;
    #pragma unroll
    for (int i = 0; i < 16; ++i) {
        const int idx = i * 256 + t;
        const int ci = idx >> 6, w = idx & 63;
        tile[ci * 65 + w] =
            x[(((size_t)b * 512 + ci0 + ci) * 64 + h) * 64 + w] * s[b * 512 + ci0 + ci];
    }
    __syncthreads();
    #pragma unroll
    for (int i = 0; i < 16; ++i) {
        const int idx = i * 256 + t;
        const int w = idx >> 6, ci = idx & 63;
        xs[(((size_t)b * 66 + h + 1) * 66 + (w + 1)) * 512 + ci0 + ci] =
            f32_to_bf16(tile[ci * 65 + w]);
    }
}

// ---------------------------------------------------------------------------
// 5) Conv as implicit GEMM on WMMA bf16, double-buffered async LDS staging.
//    Block = 256 threads (8 waves). Output tile: 128 cout x 128 pixels
//    (2 image rows x 64 cols). K pipeline: 9 taps x 16 ci-blocks = 144 steps.
//    LDS A: [g(4)][m(128)][k%8]   (linear image of pre-swizzled wt tile)
//    LDS B: [n(128)][k(32) pad 40]
// ---------------------------------------------------------------------------
__global__ __launch_bounds__(256) void conv_wmma_kernel(
    const unsigned short* __restrict__ xs,    // [8,66,66,512] bf16, halo
    const unsigned short* __restrict__ wt,    // swizzled bf16 tiles
    const float* __restrict__ demod,          // [8,512]
    const float* __restrict__ bias,           // [512]
    float* __restrict__ out)                  // [8,512,64,64] f32
{
    __shared__ unsigned short Alds[2][4096];   // 2 x 8 KB
    __shared__ unsigned short Blds[2][5120];   // 2 x 10 KB (row stride 40)

    const int t    = threadIdx.x;
    const int r0   = blockIdx.x * 2;      // image rows [r0, r0+1]
    const int coB  = blockIdx.y;          // cout tile (128 wide)
    const int co0  = coB * 128;
    const int b    = blockIdx.z;

    const int lane = t & 31;
    const int wid  = t >> 5;              // wave id 0..7 -> M sub-tile
    const int mrow = lane & 15;
    const int hi   = lane >> 4;           // 0: lanes 0-15, 1: lanes 16-31

    v8f acc[8];
    #pragma unroll
    for (int i = 0; i < 8; ++i)
        #pragma unroll
        for (int j = 0; j < 8; ++j) acc[i][j] = 0.f;

    // B-fill assignment: thread owns (pixel nB, 16-wide k-half partB)
    const int nB       = t >> 1;          // pixel 0..127
    const int partB    = t & 1;           // k 0..15 / 16..31
    const int brow_off = nB >> 6;
    const int bcol     = nB & 63;

    const int NSTEP = 144;                // 9 taps * 16 ci-blocks

    auto issue_tile = [&](int step, int buf) {
        const int kk = step >> 4;
        const int cb = step & 15;
        const int kh = kk / 3;
        const int kw = kk - kh * 3;
        // B tile: halo coords, no bounds check needed
        {
            const int hrow = r0 + brow_off + kh;   // 0..65
            const int hcol = bcol + kw;            // 0..65
            const unsigned short* src =
                xs + (((size_t)b * 66 + hrow) * 66 + hcol) * 512 + cb * 32 + partB * 16;
            unsigned short* dst = &Blds[buf][nB * 40 + partB * 16];
            copy_b128(src, dst);
            copy_b128(src + 8, dst + 8);
        }
        // A tile: linear 8KB copy of pre-swizzled weight tile
        {
            const unsigned short* src = wt + ((size_t)(step * 4 + coB)) * 4096 + t * 16;
            unsigned short* dst = &Alds[buf][t * 16];
            copy_b128(src, dst);
            copy_b128(src + 8, dst + 8);
        }
    };

    auto compute_tile = [&](int buf) {
        const unsigned short* Ab = &Alds[buf][0];
        const unsigned short* Bb = &Blds[buf][0];
        Frag16 a;
        a.u[0] = *(const uint4*)&Ab[hi * 1024 + (wid * 16 + mrow) * 8];
        a.u[1] = *(const uint4*)&Ab[2048 + hi * 1024 + (wid * 16 + mrow) * 8];
        #pragma unroll
        for (int nt = 0; nt < 8; ++nt) {
            Frag16 bf;
            const unsigned short* p = &Bb[(nt * 16 + mrow) * 40 + hi * 8];
            bf.u[0] = *(const uint4*)p;
            bf.u[1] = *(const uint4*)(p + 16);
            acc[nt] = __builtin_amdgcn_wmma_f32_16x16x32_bf16(
                false, a.v, false, bf.v, (short)0, acc[nt], false, false);
        }
    };

    issue_tile(0, 0);
    for (int step = 0; step < NSTEP; ++step) {
        const int cur = step & 1;
        if (step + 1 < NSTEP) {
            issue_tile(step + 1, cur ^ 1);   // prefetch next into other buffer
            ASYNC_WAIT(4);                   // wait for THIS step's 4 loads
        } else {
            ASYNC_WAIT(0);
        }
        __syncthreads();                     // all waves' cur-buffer data visible
        compute_tile(cur);
        __syncthreads();                     // cur buffer free for refill
    }

    // ---- epilogue: demod scale + bias, write f32 NCHW ----
    float dsc[8], bs[8];
    #pragma unroll
    for (int r = 0; r < 8; ++r) {
        const int co = co0 + wid * 16 + r + hi * 8;
        dsc[r] = demod[b * 512 + co];
        bs[r]  = bias[co];
    }
    #pragma unroll
    for (int nt = 0; nt < 8; ++nt) {
        const int pix = nt * 16 + mrow;
        const int h = r0 + (pix >> 6);
        const int w = pix & 63;
        #pragma unroll
        for (int r = 0; r < 8; ++r) {
            const int co = co0 + wid * 16 + r + hi * 8;
            out[(((size_t)b * 512 + co) * 64 + h) * 64 + w] = acc[nt][r] * dsc[r] + bs[r];
        }
    }
}

// ---------------------------------------------------------------------------
// Launch
// ---------------------------------------------------------------------------
extern "C" void kernel_launch(void* const* d_in, const int* in_sizes, int n_in,
                              void* d_out, int out_size, void* d_ws, size_t ws_size,
                              hipStream_t stream) {
    const float* x      = (const float*)d_in[0];   // [8,512,64,64]
    const float* style  = (const float*)d_in[1];   // [8,512]
    const float* weight = (const float*)d_in[2];   // [1,512,512,3,3]
    const float* bias   = (const float*)d_in[3];   // [1,512,1,1]
    const float* mod_w  = (const float*)d_in[4];   // [512,512]
    const float* mod_b  = (const float*)d_in[5];   // [512]
    float* out = (float*)d_out;

    char* ws = (char*)d_ws;
    float*          s     = (float*)(ws);                        // 16 KB
    float*          demod = (float*)(ws + 16384);                // 16 KB
    unsigned short* wt    = (unsigned short*)(ws + 32768);       // 9*512*512*2 = 4.5 MB
    unsigned short* xs    = (unsigned short*)(ws + 32768 + 9 * 512 * 512 * 2); // 8*66*66*512*2 = 34 MB

    // zero the haloed xs buffer: 8*66*66*512 halves = 2,230,272 uint4s
    zero_kernel<<<8712, 256, 0, stream>>>((uint4*)xs);
    style_linear_kernel<<<8, 256, 0, stream>>>(style, mod_w, mod_b, s);
    demod_kernel<<<dim3(2, 8), 256, 0, stream>>>(weight, s, demod);
    wtr_kernel<<<(9 * 512 * 512) / 256, 256, 0, stream>>>(weight, wt);
    xs_kernel<<<dim3(8, 64, 8), 256, 0, stream>>>(x, s, xs);
    conv_wmma_kernel<<<dim3(32, 4, 8), 256, 0, stream>>>(xs, wt, demod, bias, out);
}
